// GenerativeNet_53060025975209
// MI455X (gfx1250) — matrix operand order repaired and checked by reading, not compiled, not run on hardware
//
#include <hip/hip_runtime.h>
#include <hip/hip_bf16.h>

// ---------------------------------------------------------------------------
// MI455X (gfx1250, wave32) implementation.
// bf16 activations + v_wmma_f32_16x16x32_bf16, fp32 accumulation.
// v3: Tensor Data Mover (tensor_load_to_lds) double-buffered B-slab staging
//     (TENSORcnt + single split-barrier per K-chunk), overlapping the weight
//     DMA with WMMA compute. Falls back to sync global->LDS copies when the
//     TDM builtin is unavailable.
// ---------------------------------------------------------------------------

typedef __attribute__((ext_vector_type(16))) __bf16 v16bf;
typedef __attribute__((ext_vector_type(8)))  __bf16 v8bf;
typedef __attribute__((ext_vector_type(8)))  float  v8f;

union V16 { v16bf v; v8bf h[2]; };

#define LDSTR 40   // LDS row stride (elements): 80B = 16 data DWORDs + 4 pad DWORDs

#if defined(__has_builtin)
#if __has_builtin(__builtin_amdgcn_tensor_load_to_lds) && \
    __has_builtin(__builtin_amdgcn_s_wait_tensorcnt)
#define HAVE_TDM 1
#endif
#endif
#ifndef HAVE_TDM
#define HAVE_TDM 0
#endif

__device__ __forceinline__ unsigned short f2bf(float f) {
  union { float f; unsigned u; } c; c.f = f;
  unsigned u = c.u;
  u = u + 0x7fffu + ((u >> 16) & 1u);          // round-to-nearest-even
  return (unsigned short)(u >> 16);
}
__device__ __forceinline__ float bf2f(unsigned short h) {
  union { unsigned u; float f; } c; c.u = ((unsigned)h) << 16;
  return c.f;
}

#if HAVE_TDM
// Issue a TDM 2D tile copy: rows x 32 bf16 (row stride strideElems) from
// global -> LDS at lds_off, padding each 64B row to 80B (LDSTR elements).
__device__ __forceinline__ void tdm_issue_b(unsigned lds_off,
                                            const unsigned short* gsrc,
                                            int rows, int strideElems) {
  typedef unsigned u32x4 __attribute__((ext_vector_type(4)));
  typedef int      i32x4 __attribute__((ext_vector_type(4)));
  typedef int      i32x8 __attribute__((ext_vector_type(8)));
  unsigned long long ga = (unsigned long long)(size_t)gsrc;
  u32x4 g0;
  g0[0] = 1u;                                   // count=1 (valid user D#)
  g0[1] = lds_off;                              // lds_addr
  g0[2] = (unsigned)ga;                         // global_addr[31:0]
  g0[3] = (unsigned)(ga >> 32) | (2u << 30);    // global_addr[56:32] | type=2
  unsigned td0 = (unsigned)strideElems;         // tensor_dim0 (= stride, no OOB)
  unsigned td1 = 1u << 20;                      // tensor_dim1 (large, no OOB)
  i32x8 g1;
  g1[0] = (int)((1u << 16)      // data_size = 2 bytes
              | (1u << 20)      // pad_enable
              | (3u << 22)      // pad_interval: 16 DWORDs (one 64B row)
              | (3u << 25));    // pad_amount:   4 DWORDs (16B -> 80B stride)
  g1[1] = (int)((td0 & 0xffffu) << 16);                  // tensor_dim0 lo
  g1[2] = (int)((td0 >> 16) | ((td1 & 0xffffu) << 16));  // dim0 hi | dim1 lo
  g1[3] = (int)((td1 >> 16) | (32u << 16));              // dim1 hi | tile_dim0=32
  g1[4] = rows;                                          // tile_dim1 (tile_dim2=0)
  g1[5] = strideElems;                                   // tensor_dim0_stride
  g1[6] = 0;
  g1[7] = 0;
  i32x4 gz = {0, 0, 0, 0};
#if __clang_major__ >= 23
  i32x8 gz8 = {0, 0, 0, 0, 0, 0, 0, 0};
  __builtin_amdgcn_tensor_load_to_lds(g0, g1, gz, gz, gz8, 0);
#else
  __builtin_amdgcn_tensor_load_to_lds(g0, g1, gz, gz, 0);
#endif
}
#endif

// ---------------- conversion / weight-prep kernels -------------------------

__global__ void k_f32_to_bf16(const float* __restrict__ in,
                              unsigned short* __restrict__ out, long n) {
  long i = (long)blockIdx.x * blockDim.x + threadIdx.x;
  if (i < n) out[i] = f2bf(in[i]);
}

// Wu: [8, Cin, Cout] fp32  ->  Wut: [8*Cout, Cin] bf16 (B^T, contiguous K)
__global__ void k_prep_wu(const float* __restrict__ in,
                          unsigned short* __restrict__ out, int Cin, int Cout) {
  long i = (long)blockIdx.x * blockDim.x + threadIdx.x;
  long n = 8L * Cin * Cout;
  if (i >= n) return;
  int k = (int)(i / ((long)Cin * Cout));
  long rem = i - (long)k * Cin * Cout;
  int c = (int)(rem / Cout);
  int d = (int)(rem % Cout);
  out[((size_t)k * Cout + d) * Cin + c] = f2bf(in[i]);
}

// Wc: [27, C, C] fp32 -> Wct: [27, C, padC] bf16 transposed (d-major, c contiguous)
__global__ void k_prep_wc(const float* __restrict__ in,
                          unsigned short* __restrict__ out, int C, int padC) {
  long i = (long)blockIdx.x * blockDim.x + threadIdx.x;
  long n = 27L * C * padC;
  if (i >= n) return;
  int k = (int)(i / ((long)C * padC));
  long rem = i - (long)k * C * padC;
  int d = (int)(rem / padC);
  int c = (int)(rem % padC);
  float v = (c < C) ? in[((size_t)k * C + c) * C + d] : 0.0f;
  out[i] = f2bf(v);
}

__global__ void k_zero_stats(float* __restrict__ s, int n) {
  int i = threadIdx.x;
  for (; i < n; i += blockDim.x) s[i] = 0.0f;
}

// ---------------- WMMA GEMM: out[M,N] = A_bf16[M,K] @ Bt_bf16[N,K]^T -------
// Block: 256 threads = 8 waves, each wave one 16x(16*NT) tile (same n-range).
// B fragments staged in LDS (TDM double-buffered) and shared by all 8 waves.

template <int NT>
__global__ void k_gemm_bf16_lds(const unsigned short* __restrict__ A,
                                const unsigned short* __restrict__ Bt,
                                float* __restrict__ Out, int M, int N, int K) {
  __shared__ unsigned short sB[2][NT * 16 * LDSTR];
  const int tid  = threadIdx.x;
  const int lane = tid & 31;
  const int wave = tid >> 5;
  const int m0 = (blockIdx.x * 8 + wave) * 16;
  const int n0 = blockIdx.y * 16 * NT;
  const int hb = lane >> 4;                    // half-wave select
  const int r  = lane & 15;
  const unsigned short* arow = A + (size_t)(m0 + r) * K + hb * 8;
  v8f acc[NT];
#pragma unroll
  for (int t = 0; t < NT; ++t) acc[t] = (v8f){};

#if HAVE_TDM
  const int nch = K / 32;
  if (wave == 0)
    tdm_issue_b((unsigned)(size_t)(void*)&sB[0][0], Bt + (size_t)n0 * K, NT * 16, K);
  for (int i = 0; i < nch; ++i) {
    if (wave == 0) __builtin_amdgcn_s_wait_tensorcnt(0);
    __syncthreads();                            // DMA visible, prev buf free
    if (wave == 0 && i + 1 < nch)               // overlap next DMA with WMMAs
      tdm_issue_b((unsigned)(size_t)(void*)&sB[(i + 1) & 1][0],
                  Bt + (size_t)n0 * K + (i + 1) * 32, NT * 16, K);
    const int k0 = i * 32;
    const unsigned short* sbuf = &sB[i & 1][0];
    V16 a;
    a.h[0] = *(const v8bf*)(const void*)(arow + k0);        // K = base..base+7
    a.h[1] = *(const v8bf*)(const void*)(arow + k0 + 16);   // K = base+16..+23
#pragma unroll
    for (int t = 0; t < NT; ++t) {
      V16 b;
      const unsigned short* bp = &sbuf[(t * 16 + r) * LDSTR + hb * 8];
      b.h[0] = *(const v8bf*)(const void*)bp;
      b.h[1] = *(const v8bf*)(const void*)(bp + 16);
      acc[t] = __builtin_amdgcn_wmma_f32_16x16x32_bf16(false, a.v, false, b.v,
                                                       (short)0, acc[t], false, false);
    }
  }
#else
  for (int k0 = 0; k0 < K; k0 += 32) {
    if (tid < NT * 64) {
      const int col = tid >> 2, seg = tid & 3;
      const unsigned short* src = Bt + (size_t)(n0 + col) * K + k0 + seg * 8;
      *(v8bf*)(void*)&sB[0][col * LDSTR + seg * 8] = *(const v8bf*)(const void*)src;
    }
    __syncthreads();
    V16 a;
    a.h[0] = *(const v8bf*)(const void*)(arow + k0);
    a.h[1] = *(const v8bf*)(const void*)(arow + k0 + 16);
#pragma unroll
    for (int t = 0; t < NT; ++t) {
      V16 b;
      const unsigned short* bp = &sB[0][(t * 16 + r) * LDSTR + hb * 8];
      b.h[0] = *(const v8bf*)(const void*)bp;
      b.h[1] = *(const v8bf*)(const void*)(bp + 16);
      acc[t] = __builtin_amdgcn_wmma_f32_16x16x32_bf16(false, a.v, false, b.v,
                                                       (short)0, acc[t], false, false);
    }
    __syncthreads();
  }
#endif
#pragma unroll
  for (int t = 0; t < NT; ++t) {
    float* out = Out + (size_t)(m0 + hb * 8) * N + n0 + t * 16 + r;
#pragma unroll
    for (int j = 0; j < 8; ++j) out[(size_t)j * N] = acc[t][j];
  }
}

// -------- WMMA sparse conv: out[M,C] = sum_k X[nbr[k]] @ Wc[k] -------------
// X: [M+1, padC] bf16 (row M = zeros for missing neighbors)
// Wt: [27, C, padC] (d-major, c contiguous). B staged in LDS via TDM.

template <int NT>
__global__ void k_sconv_wmma(const unsigned short* __restrict__ X,
                             const int* __restrict__ nbr,
                             const unsigned short* __restrict__ Wt,
                             float* __restrict__ Out, int M, int C, int padC) {
  __shared__ unsigned short sB[2][NT * 16 * LDSTR];
  const int tid  = threadIdx.x;
  const int lane = tid & 31;
  const int wave = tid >> 5;
  const int m0 = (blockIdx.x * 8 + wave) * 16;
  const int n0 = blockIdx.y * 16 * NT;
  const int hb = lane >> 4;
  const int r  = lane & 15;
  v8f acc[NT];
#pragma unroll
  for (int t = 0; t < NT; ++t) acc[t] = (v8f){};

#if HAVE_TDM
  const int cpk = padC / 32;                   // K-chunks per tap
  const int nch = 27 * cpk;
  if (wave == 0)
    tdm_issue_b((unsigned)(size_t)(void*)&sB[0][0],
                Wt + (size_t)n0 * padC, NT * 16, padC);
  int chunk = 0;
  for (int k = 0; k < 27; ++k) {
    const int src = nbr[(size_t)k * M + m0 + r];          // src==M -> zero row
    const unsigned short* arow = X + (size_t)src * padC + hb * 8;
    if (k + 1 < 27) {                                     // warm next gather
      int nxt = nbr[(size_t)(k + 1) * M + m0 + r];
      __builtin_prefetch(X + (size_t)nxt * padC, 0, 0);
    }
    for (int j = 0; j < cpk; ++j, ++chunk) {
      if (wave == 0) __builtin_amdgcn_s_wait_tensorcnt(0);
      __syncthreads();
      if (wave == 0 && chunk + 1 < nch) {
        const int kn = (chunk + 1) / cpk, jn = (chunk + 1) % cpk;
        tdm_issue_b((unsigned)(size_t)(void*)&sB[(chunk + 1) & 1][0],
                    Wt + ((size_t)kn * C + n0) * padC + jn * 32, NT * 16, padC);
      }
      const int c0 = j * 32;
      const unsigned short* sbuf = &sB[chunk & 1][0];
      V16 a;
      a.h[0] = *(const v8bf*)(const void*)(arow + c0);
      a.h[1] = *(const v8bf*)(const void*)(arow + c0 + 16);
#pragma unroll
      for (int t = 0; t < NT; ++t) {
        V16 b;
        const unsigned short* bp = &sbuf[(t * 16 + r) * LDSTR + hb * 8];
        b.h[0] = *(const v8bf*)(const void*)bp;
        b.h[1] = *(const v8bf*)(const void*)(bp + 16);
        acc[t] = __builtin_amdgcn_wmma_f32_16x16x32_bf16(false, a.v, false, b.v,
                                                         (short)0, acc[t], false, false);
      }
    }
  }
#else
  for (int k = 0; k < 27; ++k) {
    const int src = nbr[(size_t)k * M + m0 + r];
    const unsigned short* arow = X + (size_t)src * padC + hb * 8;
    if (k + 1 < 27) {
      int nxt = nbr[(size_t)(k + 1) * M + m0 + r];
      __builtin_prefetch(X + (size_t)nxt * padC, 0, 0);
    }
    for (int c0 = 0; c0 < padC; c0 += 32) {
      if (tid < NT * 64) {
        const int col = tid >> 2, seg = tid & 3;
        const unsigned short* wsrc =
            Wt + ((size_t)k * C + n0 + col) * padC + c0 + seg * 8;
        *(v8bf*)(void*)&sB[0][col * LDSTR + seg * 8] = *(const v8bf*)(const void*)wsrc;
      }
      __syncthreads();
      V16 a;
      a.h[0] = *(const v8bf*)(const void*)(arow + c0);
      a.h[1] = *(const v8bf*)(const void*)(arow + c0 + 16);
#pragma unroll
      for (int t = 0; t < NT; ++t) {
        V16 b;
        const unsigned short* bp = &sB[0][(t * 16 + r) * LDSTR + hb * 8];
        b.h[0] = *(const v8bf*)(const void*)bp;
        b.h[1] = *(const v8bf*)(const void*)(bp + 16);
        acc[t] = __builtin_amdgcn_wmma_f32_16x16x32_bf16(false, a.v, false, b.v,
                                                         (short)0, acc[t], false, false);
      }
      __syncthreads();
    }
  }
#endif
#pragma unroll
  for (int t = 0; t < NT; ++t) {
    float* out = Out + (size_t)(m0 + hb * 8) * C + n0 + t * 16 + r;
#pragma unroll
    for (int j = 0; j < 8; ++j) out[(size_t)j * C] = acc[t][j];
  }
}

// ---------------- BatchNorm: column sums / apply + ELU ---------------------

__global__ void k_bn_reduce(const float* __restrict__ x, float* __restrict__ stats,
                            int N, int C) {
  int col = blockIdx.y * blockDim.x + threadIdx.x;
  if (col >= C) return;
  const float* p = x + (size_t)blockIdx.x * 1024 * C + col;
  float s = 0.0f, s2 = 0.0f;
  for (int r0 = 0; r0 < 1024; ++r0) {
    float v = p[(size_t)r0 * C];
    s += v; s2 += v * v;
  }
  atomicAdd(&stats[col], s);
  atomicAdd(&stats[C + col], s2);
}

__global__ void k_bn_apply_elu(const float* __restrict__ x,
                               const float* __restrict__ stats,
                               const float* __restrict__ gamma,
                               const float* __restrict__ beta,
                               unsigned short* __restrict__ out,
                               int N, int C, int stride, int zeroRow, float invN) {
  long i = (long)blockIdx.x * blockDim.x + threadIdx.x;
  long total = (long)(N + zeroRow) * stride;
  if (i >= total) return;
  int row = (int)(i / stride);
  int col = (int)(i % stride);
  if (row >= N || col >= C) { out[i] = 0; return; }
  float mean = stats[col] * invN;
  float var  = stats[C + col] * invN - mean * mean;
  float sc   = gamma[col] * rsqrtf(var + 1e-5f);
  float sh   = beta[col] - mean * sc;
  float v = x[(size_t)row * C + col] * sc + sh;
  v = (v > 0.0f) ? v : expm1f(v);                         // ELU(alpha=1)
  out[i] = f2bf(v);
}

// ---------------- classifier / prune / gather ------------------------------

__global__ void k_classifier(const unsigned short* __restrict__ x,
                             const float* __restrict__ w,
                             const float* __restrict__ bias,
                             float* __restrict__ cls, int N, int C) {
  int wave = threadIdx.x >> 5, lane = threadIdx.x & 31;
  int row = blockIdx.x * 8 + wave;
  if (row >= N) return;
  const unsigned short* p = x + (size_t)row * C;
  float s = 0.0f;
  for (int c = lane; c < C; c += 32) s += bf2f(p[c]) * w[c];
#pragma unroll
  for (int o = 16; o > 0; o >>= 1) s += __shfl_xor(s, o, 32);
  if (lane == 0) cls[row] = s + bias[0];
}

// masked survivor gather: out[i,:] = (cls[sel[i]]>0) ? x[sel[i],:] : 0
__global__ void k_mask_gather(const unsigned short* __restrict__ x,
                              const float* __restrict__ cls,
                              const int* __restrict__ sel,
                              unsigned short* __restrict__ out, int Nsel, int C) {
  long i = (long)blockIdx.x * blockDim.x + threadIdx.x;
  long total = (long)Nsel * C;
  if (i >= total) return;
  int r = (int)(i / C), c = (int)(i % C);
  int s = sel[r];
  out[i] = (cls[s] > 0.0f) ? x[(size_t)s * C + c] : (unsigned short)0;
}

// final masked fp32 output y = (cls>0) * y   (C = 16)
__global__ void k_final_out(const unsigned short* __restrict__ x,
                            const float* __restrict__ cls,
                            float* __restrict__ y, long N) {
  long i = (long)blockIdx.x * blockDim.x + threadIdx.x;
  if (i >= N * 16) return;
  long row = i >> 4;
  y[i] = (cls[row] > 0.0f) ? bf2f(x[i]) : 0.0f;
}

__global__ void k_targets_copy(const unsigned char* __restrict__ t,
                               float* __restrict__ out, long n) {
  long i = (long)blockIdx.x * blockDim.x + threadIdx.x;
  if (i < n) out[i] = t[i] ? 1.0f : 0.0f;
}

// ---------------------------------------------------------------------------

extern "C" void kernel_launch(void* const* d_in, const int* in_sizes, int n_in,
                              void* d_out, int out_size, void* d_ws, size_t ws_size,
                              hipStream_t stream) {
  (void)in_sizes; (void)n_in; (void)out_size; (void)ws_size;

  static const int CIN[7]  = {512, 1024, 512, 256, 128, 64, 32};
  static const int COUT[7] = {1024, 512, 256, 128, 64, 32, 16};
  static const int NINA[7] = {512, 4096, 8192, 16384, 32768, 65536, 131072};
  static const int SELN[5] = {8192, 16384, 32768, 65536, 131072};
  static const long BON[6] = {32768, 65536, 131072, 262144, 524288, 1048576};

  // workspace carve (256B aligned regions)
  size_t off = 0;
  auto carve = [&](size_t bytes) -> char* {
    char* p = (char*)d_ws + off;
    off += (bytes + 255) & ~(size_t)255;
    return p;
  };
  float*          bufF  = (float*)carve(16777216ull * 4);           // fp32 GEMM out (reused)
  unsigned short* bufGU = (unsigned short*)carve(33554464ull * 2);  // conv input (+zero row, padded)
  unsigned short* bufX  = (unsigned short*)carve(16777216ull * 2);  // post-BN2 activations
  unsigned short* bufS  = (unsigned short*)carve(4194304ull * 2);   // gathered survivors / z
  unsigned short* wut   = (unsigned short*)carve(4194304ull * 2);   // bf16 Wu^T
  unsigned short* wct   = (unsigned short*)carve(28311552ull * 2);  // bf16 Wc^T (padded)
  float*          stats = (float*)carve(2048 * 4);

  // d_out layout: cls[0..5] | targets[0..5] | y_final
  size_t clsOff[6]; clsOff[0] = 0;
  for (int i = 1; i < 6; ++i) clsOff[i] = clsOff[i - 1] + BON[i - 1];
  size_t tgtBase = clsOff[5] + BON[5];       // 2,064,384
  size_t yOff    = tgtBase * 2;              // 4,128,768
  float* outF = (float*)d_out;

  // z -> bf16
  k_f32_to_bf16<<<(262144 + 255) / 256, 256, 0, stream>>>(
      (const float*)d_in[0], bufS, 262144L);
  const unsigned short* curA = bufS;

  int si = 0;
  for (int blk = 0; blk < 6; ++blk) {
    int nsub = (blk == 0) ? 2 : 1;
    for (int s = 0; s < nsub; ++s, ++si) {
      const int Cin = CIN[si], Cout = COUT[si], Nin = NINA[si];
      const int Nout = 8 * Nin;
      const int padC = (Cout < 32) ? 32 : Cout;
      const float* Wu = (const float*)d_in[1 + 6 * si + 0];
      const float* g1 = (const float*)d_in[1 + 6 * si + 1];
      const float* b1 = (const float*)d_in[1 + 6 * si + 2];
      const float* Wc = (const float*)d_in[1 + 6 * si + 3];
      const float* g2 = (const float*)d_in[1 + 6 * si + 4];
      const float* b2 = (const float*)d_in[1 + 6 * si + 5];
      const int*  nbr = (const int*)d_in[55 + si];

      // weight repack to bf16 / transposed
      long nwu = 8L * Cin * Cout;
      k_prep_wu<<<(unsigned)((nwu + 255) / 256), 256, 0, stream>>>(Wu, wut, Cin, Cout);
      long nwc = 27L * Cout * padC;
      k_prep_wc<<<(unsigned)((nwc + 255) / 256), 256, 0, stream>>>(Wc, wct, Cout, padC);

      // generative up: [Nin,Cin] x [Cin,8*Cout] -> bufF viewed as [Nout,Cout]
      // N = 8*Cout >= 128 -> always NT=4 (grid.y = N/64)
      dim3 gU(Nin / 128, (8 * Cout) / 64);
      k_gemm_bf16_lds<4><<<gU, 256, 0, stream>>>(curA, wut, bufF, Nin, 8 * Cout, Cin);

      // BN1 + ELU -> bufGU (padded K stride, zero row at Nout)
      k_zero_stats<<<1, 256, 0, stream>>>(stats, 2 * Cout);
      dim3 gR(Nout / 1024, (Cout + 255) / 256);
      k_bn_reduce<<<gR, 256, 0, stream>>>(bufF, stats, Nout, Cout);
      long ngu = (long)(Nout + 1) * padC;
      k_bn_apply_elu<<<(unsigned)((ngu + 255) / 256), 256, 0, stream>>>(
          bufF, stats, g1, b1, bufGU, Nout, Cout, padC, 1, 1.0f / (float)Nout);

      // sparse 3^3 conv (gather-GEMM over 27 taps), NT by channel width
      if (Cout >= 64) {
        dim3 gC(Nout / 128, Cout / 64);
        k_sconv_wmma<4><<<gC, 256, 0, stream>>>(bufGU, nbr, wct, bufF, Nout, Cout, padC);
      } else if (Cout == 32) {
        dim3 gC(Nout / 128, 1);
        k_sconv_wmma<2><<<gC, 256, 0, stream>>>(bufGU, nbr, wct, bufF, Nout, Cout, padC);
      } else {
        dim3 gC(Nout / 128, 1);
        k_sconv_wmma<1><<<gC, 256, 0, stream>>>(bufGU, nbr, wct, bufF, Nout, Cout, padC);
      }

      // BN2 + ELU -> bufX
      k_zero_stats<<<1, 256, 0, stream>>>(stats, 2 * Cout);
      k_bn_reduce<<<gR, 256, 0, stream>>>(bufF, stats, Nout, Cout);
      long nx = (long)Nout * Cout;
      k_bn_apply_elu<<<(unsigned)((nx + 255) / 256), 256, 0, stream>>>(
          bufF, stats, g2, b2, bufX, Nout, Cout, Cout, 0, 1.0f / (float)Nout);
      curA = bufX;
    }

    // classifier + prune for this block
    const int C = COUT[si - 1];
    const long N = BON[blk];
    const float* cw = (const float*)d_in[43 + 2 * blk];
    const float* cb = (const float*)d_in[44 + 2 * blk];
    float* clsPtr = outF + clsOff[blk];
    k_classifier<<<(unsigned)((N + 7) / 8), 256, 0, stream>>>(bufX, cw, cb, clsPtr, (int)N, C);

    if (blk < 5) {
      const int* sel = (const int*)d_in[62 + blk];
      long ns = (long)SELN[blk] * C;
      k_mask_gather<<<(unsigned)((ns + 255) / 256), 256, 0, stream>>>(
          bufX, clsPtr, sel, bufS, SELN[blk], C);
      curA = bufS;
    } else {
      long ny = N * 16;
      k_final_out<<<(unsigned)((ny + 255) / 256), 256, 0, stream>>>(
          bufX, clsPtr, outF + yOff, N);
    }
  }

  // pass-through targets
  size_t toff = tgtBase;
  for (int blk = 0; blk < 6; ++blk) {
    k_targets_copy<<<(unsigned)((BON[blk] + 255) / 256), 256, 0, stream>>>(
        (const unsigned char*)d_in[67 + blk], outF + toff, BON[blk]);
    toff += BON[blk];
  }
}
